// RQVAE_9423158247848
// MI455X (gfx1250) — matrix operand order repaired and checked by reading, not compiled, not run on hardware
//
#include <hip/hip_runtime.h>
#include <hip/hip_bf16.h>
#include <math.h>

// ---------------------------------------------------------------------------
// Problem constants (from reference)
// ---------------------------------------------------------------------------
constexpr int cB   = 16384;
constexpr int cIN  = 1024;
constexpr int cHID = 2048;
constexpr int cLAT = 256;
constexpr int cL   = 4;
constexpr int cK   = 4096;

// ---------------------------------------------------------------------------
// WMMA types (gfx1250, wave32)
// ---------------------------------------------------------------------------
typedef __attribute__((ext_vector_type(16))) __bf16 v16bf;
typedef __attribute__((ext_vector_type(8)))  float  v8f;

union ABfrag {
    uint4 q[2];   // 2 x 16B loads
    v16bf v;      // fed to v_wmma_f32_16x16x32_bf16
};

__device__ __forceinline__ unsigned short f32_to_bf16(float f) {
    unsigned int u = __float_as_uint(f);
    unsigned int r = u + 0x7FFFu + ((u >> 16) & 1u);  // round-to-nearest-even
    return (unsigned short)(r >> 16);
}

// ---------------------------------------------------------------------------
// CDNA5 async global->LDS copy (ASYNCcnt-tracked DMA, no VGPR staging).
// Per lane: LDS[lds_addr] = MEM[gaddr], 16 bytes.
// ---------------------------------------------------------------------------
__device__ __forceinline__ void async_copy_b128(unsigned lds_addr,
                                                const unsigned short* gaddr) {
    asm volatile("global_load_async_to_lds_b128 %0, %1, off"
                 :: "v"(lds_addr), "v"((unsigned long long)(size_t)gaddr)
                 : "memory");
}

__device__ __forceinline__ void wait_async0() {
    asm volatile("s_wait_asynccnt 0x0" ::: "memory");
}

// ---------------------------------------------------------------------------
// fp32 -> bf16 conversion (grid-stride)
// ---------------------------------------------------------------------------
__global__ void cvt_bf16_kernel(const float* __restrict__ in,
                                unsigned short* __restrict__ out, size_t n) {
    size_t i = (size_t)blockIdx.x * blockDim.x + threadIdx.x;
    size_t stride = (size_t)gridDim.x * blockDim.x;
    for (; i < n; i += stride) out[i] = f32_to_bf16(in[i]);
}

// fp32 [R][C] -> bf16 transposed [C][R], 32x32 LDS tiles
__global__ __launch_bounds__(256) void cvt_transpose_kernel(
    const float* __restrict__ in, unsigned short* __restrict__ out,
    int R, int Ccol) {
    __shared__ float tile[32][33];
    const int r0 = blockIdx.y * 32, c0 = blockIdx.x * 32;
    const int tx = threadIdx.x & 31, ty = threadIdx.x >> 5;  // 8 rows of 32
#pragma unroll
    for (int i = ty; i < 32; i += 8)
        tile[i][tx] = in[(size_t)(r0 + i) * Ccol + c0 + tx];
    __syncthreads();
#pragma unroll
    for (int i = ty; i < 32; i += 8)
        out[(size_t)(c0 + i) * R + r0 + tx] = f32_to_bf16(tile[tx][i]);
}

__global__ void zero_kernel(float* __restrict__ p, size_t n) {
    size_t i = (size_t)blockIdx.x * blockDim.x + threadIdx.x;
    size_t stride = (size_t)gridDim.x * blockDim.x;
    for (; i < n; i += stride) p[i] = 0.0f;
}

// per-code squared L2 norm of codebooks [L*K, LAT]
__global__ void cnorm_kernel(const float* __restrict__ cb,
                             float* __restrict__ cnorm, int ncodes, int dim) {
    int i = blockIdx.x * blockDim.x + threadIdx.x;
    if (i >= ncodes) return;
    const float* p = cb + (size_t)i * dim;
    float s = 0.f;
    for (int c = 0; c < dim; ++c) s += p[c] * p[c];
    cnorm[i] = s;
}

// ---------------------------------------------------------------------------
// Tiled bf16 WMMA GEMM:  C[M,N] = A[M,K] @ Wt[N,K]^T + bias[N]
// A row-major [M][K]; Wt is the PRE-TRANSPOSED weight [N][K].
// Block tile 128x128, BK=64, async-DMA double-buffered LDS,
// 256 thr = 8 waves (4x2), per-wave 32x64, 16 WMMAs per staged tile.
// ---------------------------------------------------------------------------
#define GBM 128
#define GBN 128
#define GBK 64
#define GLD 72   // padded LDS stride (elems): 144B rows, conflict-free b128

__global__ __launch_bounds__(256) void gemm_bf16_kernel(
    const unsigned short* __restrict__ A, const unsigned short* __restrict__ Wt,
    const float* __restrict__ bias, float* __restrict__ C,
    int M, int N, int K)
{
    __shared__ alignas(16) unsigned short As[2][GBM * GLD];
    __shared__ alignas(16) unsigned short Bs[2][GBN * GLD];

    const int tid  = threadIdx.x;
    const int lane = tid & 31;
    const int wave = tid >> 5;
    const int wm   = wave >> 1;    // 0..3 -> 32 rows each
    const int wn   = wave & 1;     // 0..1 -> 64 cols each
    const int half = lane >> 4;
    const int l16  = lane & 15;

    const int bm0 = blockIdx.y * GBM;
    const int bn0 = blockIdx.x * GBN;

    // loader mapping: 2 threads per row, each 32 contiguous bf16 (64B)
    const int lrow = tid >> 1;          // 0..127
    const int lcol = (tid & 1) * 32;    // 0 or 32

    const unsigned short* ga = A  + (size_t)(bm0 + lrow) * K + lcol;
    const unsigned short* gb = Wt + (size_t)(bn0 + lrow) * K + lcol;

    // async-DMA stage of one BK tile into LDS buffer `buf`
    auto issue_tile = [&](int buf, int kt) {
        const unsigned short* pa = ga + (size_t)kt * GBK;
        const unsigned short* pb = gb + (size_t)kt * GBK;
        unsigned la = (unsigned)(size_t)&As[buf][lrow * GLD + lcol];
        unsigned lb = (unsigned)(size_t)&Bs[buf][lrow * GLD + lcol];
#pragma unroll
        for (int i = 0; i < 4; ++i) {
            async_copy_b128(la + 16u * i, pa + 8 * i);
            async_copy_b128(lb + 16u * i, pb + 8 * i);
        }
    };

    v8f acc[2][4];
#pragma unroll
    for (int i = 0; i < 2; ++i)
#pragma unroll
        for (int j = 0; j < 4; ++j) acc[i][j] = v8f{0.f,0.f,0.f,0.f,0.f,0.f,0.f,0.f};

    issue_tile(0, 0);
    wait_async0();
    __syncthreads();

    const int KT = K / GBK;
    for (int kt = 0; kt < KT; ++kt) {
        const int cur = kt & 1;
        const int nxt = cur ^ 1;
        if (kt + 1 < KT) issue_tile(nxt, kt + 1);   // DMA overlaps compute

        // 2 k-steps of 32 from the current LDS buffer
#pragma unroll
        for (int kk = 0; kk < 2; ++kk) {
            ABfrag af[2], bf[4];
#pragma unroll
            for (int mt = 0; mt < 2; ++mt) {
                const unsigned short* p =
                    &As[cur][(wm * 32 + mt * 16 + l16) * GLD + kk * 32 + half * 8];
                af[mt].q[0] = *(const uint4*)(p);        // K = half*8 + 0..7
                af[mt].q[1] = *(const uint4*)(p + 16);   // K = 16 + half*8 + 0..7
            }
#pragma unroll
            for (int nt = 0; nt < 4; ++nt) {
                const unsigned short* p =
                    &Bs[cur][(wn * 64 + nt * 16 + l16) * GLD + kk * 32 + half * 8];
                bf[nt].q[0] = *(const uint4*)(p);        // K = half*8 + 0..7
                bf[nt].q[1] = *(const uint4*)(p + 16);   // K = 16 + half*8 + 0..7
            }
#pragma unroll
            for (int mt = 0; mt < 2; ++mt)
#pragma unroll
                for (int nt = 0; nt < 4; ++nt)
                    acc[mt][nt] = __builtin_amdgcn_wmma_f32_16x16x32_bf16(
                        false, af[mt].v, false, bf[nt].v, (short)0, acc[mt][nt],
                        false, false);
        }
        wait_async0();     // own fills of nxt landed (had whole compute to finish)
        __syncthreads();   // => all waves' fills done + all reads of cur done
    }

    // ---- epilogue: C = acc + bias ----
#pragma unroll
    for (int mt = 0; mt < 2; ++mt) {
#pragma unroll
        for (int nt = 0; nt < 4; ++nt) {
            int n = bn0 + wn * 64 + nt * 16 + l16;
            float bv = bias[n];
#pragma unroll
            for (int r = 0; r < 8; ++r) {
                int m = bm0 + wm * 32 + mt * 16 + r + half * 8;
                C[(size_t)m * N + n] = acc[mt][nt][r] + bv;
            }
        }
    }
}

// ---------------------------------------------------------------------------
// LayerNorm + exact GELU, fp32 in -> bf16 out. One block (256 thr) per row.
// ---------------------------------------------------------------------------
__global__ __launch_bounds__(256) void ln_gelu_kernel(
    const float* __restrict__ X, const float* __restrict__ g,
    const float* __restrict__ be, unsigned short* __restrict__ Y, int Ncols)
{
    __shared__ float red[256];
    const int row = blockIdx.x;
    const float* x = X + (size_t)row * Ncols;

    float lsum = 0.f;
    for (int c = threadIdx.x; c < Ncols; c += 256) lsum += x[c];
    red[threadIdx.x] = lsum; __syncthreads();
    for (int s = 128; s > 0; s >>= 1) {
        if (threadIdx.x < s) red[threadIdx.x] += red[threadIdx.x + s];
        __syncthreads();
    }
    float mean = red[0] / (float)Ncols;
    __syncthreads();

    float lvar = 0.f;
    for (int c = threadIdx.x; c < Ncols; c += 256) {
        float d = x[c] - mean; lvar += d * d;
    }
    red[threadIdx.x] = lvar; __syncthreads();
    for (int s = 128; s > 0; s >>= 1) {
        if (threadIdx.x < s) red[threadIdx.x] += red[threadIdx.x + s];
        __syncthreads();
    }
    float rstd = rsqrtf(red[0] / (float)Ncols + 1e-5f);

    unsigned short* y = Y + (size_t)row * Ncols;
    for (int c = threadIdx.x; c < Ncols; c += 256) {
        float v  = (x[c] - mean) * rstd * g[c] + be[c];
        float ge = 0.5f * v * (1.0f + erff(v * 0.70710678118654752f));
        y[c] = f32_to_bf16(ge);
    }
}

// ---------------------------------------------------------------------------
// Residual-VQ level: WMMA distance GEMM + argmin + residual/qsum update.
// Block = 128 threads (4 waves); each wave owns a 16-row tile.
// Codebook tiles (16 codes x 256) are async-DMA staged into LDS once per
// block, double-buffered, shared by all 4 waves.
// dist = ||c||^2 - 2 r.c  (||r||^2 constant per row, dropped for argmin)
// ---------------------------------------------------------------------------
#define CLD 264   // padded stride: 528B rows -> conflict-free 16-lane b128

__global__ __launch_bounds__(128) void vq_level_kernel(
    int level,
    float* __restrict__ residual,              // [B,LAT] in/out
    float* __restrict__ qsum,                  // [B,LAT] accumulate
    const unsigned short* __restrict__ cbbf,   // [L,K,LAT] bf16
    const float* __restrict__ cbf32,           // [L,K,LAT] fp32
    const float* __restrict__ cnorm,           // [L,K]
    float* __restrict__ out_ids,               // [B,L] (as float)
    float* __restrict__ commit_accum)          // scalar accumulator
{
    __shared__ alignas(16) unsigned short rbf[4][16 * CLD];   // residual tiles
    __shared__ alignas(16) unsigned short cbuf[2][16 * CLD];  // codebook tile
    __shared__ int sidx[4][16];

    const int tid  = threadIdx.x;
    const int lane = tid & 31;
    const int wave = tid >> 5;
    const int half = lane >> 4;
    const int l16  = lane & 15;
    const int rowBase = (blockIdx.x * 4 + wave) * 16;

    const unsigned short* cb = cbbf + (size_t)level * cK * cLAT;
    const float* cn = cnorm + (size_t)level * cK;

    // cooperative codebook stage mapping: 8 threads per code row, 32 elems each
    const int crow = tid >> 3;         // 0..15
    const int ccol = (tid & 7) * 32;   // 0..224

    auto issue_cb = [&](int buf, int nt2) {
        const unsigned short* src = cb + (size_t)(nt2 * 16 + crow) * cLAT + ccol;
        unsigned lc = (unsigned)(size_t)&cbuf[buf][crow * CLD + ccol];
#pragma unroll
        for (int i = 0; i < 4; ++i) async_copy_b128(lc + 16u * i, src + 8 * i);
    };

    issue_cb(0, 0);   // DMA tile 0 while we stage the residual

    // stage residual tile (16 x LAT) into LDS as bf16
    unsigned short* myr = rbf[wave];
    for (int i = lane; i < 16 * cLAT; i += 32) {
        int r = i >> 8;        // /256
        int c = i & (cLAT - 1);
        myr[r * CLD + c] = f32_to_bf16(residual[(size_t)(rowBase + r) * cLAT + c]);
    }
    wait_async0();
    __syncthreads();

    // preload all 8 A fragments (K = 256 = 8 * 32)
    ABfrag af[8];
#pragma unroll
    for (int kk = 0; kk < 8; ++kk) {
        const unsigned short* p = &myr[l16 * CLD + kk * 32 + half * 8];
        af[kk].q[0] = *(const uint4*)(p);
        af[kk].q[1] = *(const uint4*)(p + 16);
    }

    float minv[8]; int mini[8];
#pragma unroll
    for (int r = 0; r < 8; ++r) { minv[r] = 3.4e38f; mini[r] = 0; }

    const int NT = cK / 16;
    for (int nt = 0; nt < NT; ++nt) {
        const int cur = nt & 1;
        const int nxt = cur ^ 1;
        if (nt + 1 < NT) issue_cb(nxt, nt + 1);   // DMA next tile during compute

        v8f acc = v8f{0.f,0.f,0.f,0.f,0.f,0.f,0.f,0.f};
        const unsigned short* cp = &cbuf[cur][l16 * CLD + half * 8];
#pragma unroll
        for (int kk = 0; kk < 8; ++kk) {
            ABfrag bfr;
            bfr.q[0] = *(const uint4*)(cp + kk * 32);
            bfr.q[1] = *(const uint4*)(cp + kk * 32 + 16);
            acc = __builtin_amdgcn_wmma_f32_16x16x32_bf16(
                false, af[kk].v, false, bfr.v, (short)0, acc, false, false);
        }
        const int code = nt * 16 + l16;
        const float cnv = cn[code];
#pragma unroll
        for (int r = 0; r < 8; ++r) {
            float d = cnv - 2.0f * acc[r];
            if (d < minv[r]) { minv[r] = d; mini[r] = code; }
        }
        wait_async0();
        __syncthreads();
    }

    // reduce argmin across the 16 lanes of each half-group (first-index ties)
#pragma unroll
    for (int r = 0; r < 8; ++r) {
#pragma unroll
        for (int off = 8; off >= 1; off >>= 1) {
            float ov = __shfl_xor(minv[r], off, 16);
            int   oi = __shfl_xor(mini[r], off, 16);
            if (ov < minv[r] || (ov == minv[r] && oi < mini[r])) {
                minv[r] = ov; mini[r] = oi;
            }
        }
    }
    if (l16 == 0) {
#pragma unroll
        for (int r = 0; r < 8; ++r) {
            int row = half * 8 + r;
            sidx[wave][row] = mini[r];
            out_ids[(size_t)(rowBase + row) * cL + level] = (float)mini[r];
        }
    }
    __syncthreads();

    // update residual, qsum, commit partial (fp32)
    const float* cbl = cbf32 + (size_t)level * cK * cLAT;
    float csum = 0.f;
    for (int r = 0; r < 16; ++r) {
        const int idx = sidx[wave][r];
        const float* q = cbl + (size_t)idx * cLAT;
        const size_t base = (size_t)(rowBase + r) * cLAT;
        for (int c = lane; c < cLAT; c += 32) {
            float qv = q[c];
            float rv = residual[base + c];
            float dd = rv - qv;
            csum += dd * dd;
            residual[base + c] = dd;
            qsum[base + c] += qv;
        }
    }
    for (int off = 16; off >= 1; off >>= 1) csum += __shfl_xor(csum, off, 32);
    if (lane == 0) atomicAdd(commit_accum, csum);
}

// ---------------------------------------------------------------------------
// sum((a-b)^2) reduction -> atomicAdd into *accum
// ---------------------------------------------------------------------------
__global__ __launch_bounds__(256) void sqdiff_kernel(
    const float* __restrict__ a, const float* __restrict__ b,
    float* __restrict__ accum, size_t n)
{
    __shared__ float red[256];
    size_t i = (size_t)blockIdx.x * blockDim.x + threadIdx.x;
    size_t stride = (size_t)gridDim.x * blockDim.x;
    float s = 0.f;
    for (; i < n; i += stride) { float d = a[i] - b[i]; s += d * d; }
    red[threadIdx.x] = s; __syncthreads();
    for (int k = 128; k > 0; k >>= 1) {
        if (threadIdx.x < k) red[threadIdx.x] += red[threadIdx.x + k];
        __syncthreads();
    }
    if (threadIdx.x == 0) atomicAdd(accum, red[0]);
}

__global__ void finalize_kernel(const float* __restrict__ loss, float* __restrict__ out) {
    float recon  = loss[1] / (float)((size_t)cB * cIN);
    float commit = 0.25f * loss[0] / (float)((size_t)cB * cLAT);
    out[0] = recon;
    out[1] = commit;
    out[2] = recon + commit;
}

// ---------------------------------------------------------------------------
// host launch
// ---------------------------------------------------------------------------
extern "C" void kernel_launch(void* const* d_in, const int* in_sizes, int n_in,
                              void* d_out, int out_size, void* d_ws, size_t ws_size,
                              hipStream_t stream) {
    (void)in_sizes; (void)n_in; (void)out_size; (void)ws_size;

    const float* x       = (const float*)d_in[0];
    const float* enc_w1  = (const float*)d_in[1];
    const float* enc_b1  = (const float*)d_in[2];
    const float* enc_g1  = (const float*)d_in[3];
    const float* enc_be1 = (const float*)d_in[4];
    const float* enc_w2  = (const float*)d_in[5];
    const float* enc_b2  = (const float*)d_in[6];
    const float* enc_g2  = (const float*)d_in[7];
    const float* enc_be2 = (const float*)d_in[8];
    const float* enc_w3  = (const float*)d_in[9];
    const float* enc_b3  = (const float*)d_in[10];
    const float* cbooks  = (const float*)d_in[11];
    const float* dec_w1  = (const float*)d_in[12];
    const float* dec_b1  = (const float*)d_in[13];
    const float* dec_g1  = (const float*)d_in[14];
    const float* dec_be1 = (const float*)d_in[15];
    const float* dec_w2  = (const float*)d_in[16];
    const float* dec_b2  = (const float*)d_in[17];
    const float* dec_g2  = (const float*)d_in[18];
    const float* dec_be2 = (const float*)d_in[19];
    const float* dec_w3  = (const float*)d_in[20];
    const float* dec_b3  = (const float*)d_in[21];

    char* ws = (char*)d_ws;
    size_t off = 0;
    auto alloc = [&](size_t bytes) {
        size_t o = off; off += (bytes + 255) & ~(size_t)255; return o;
    };
    unsigned short* xbf    = (unsigned short*)(ws + alloc((size_t)cB * cIN * 2));
    unsigned short* w1t    = (unsigned short*)(ws + alloc((size_t)cIN * cHID * 2));
    unsigned short* w2t    = (unsigned short*)(ws + alloc((size_t)cHID * cHID * 2));
    unsigned short* w3t    = (unsigned short*)(ws + alloc((size_t)cHID * cLAT * 2));
    unsigned short* dw1t   = (unsigned short*)(ws + alloc((size_t)cLAT * cHID * 2));
    unsigned short* dw2t   = (unsigned short*)(ws + alloc((size_t)cHID * cHID * 2));
    unsigned short* dw3t   = (unsigned short*)(ws + alloc((size_t)cHID * cIN * 2));
    unsigned short* cbbf   = (unsigned short*)(ws + alloc((size_t)cL * cK * cLAT * 2));
    float*          cnorm  = (float*)(ws + alloc((size_t)cL * cK * 4));
    float*          gbuf   = (float*)(ws + alloc((size_t)cB * cHID * 4));
    unsigned short* act    = (unsigned short*)(ws + alloc((size_t)cB * cHID * 2));
    float*          res    = (float*)(ws + alloc((size_t)cB * cLAT * 4));
    float*          qsum   = (float*)(ws + alloc((size_t)cB * cLAT * 4));
    unsigned short* qsumbf = (unsigned short*)(ws + alloc((size_t)cB * cLAT * 2));
    float*          loss   = (float*)(ws + alloc(256));   // [0]=commit sum, [1]=recon sum

    float* out_f   = (float*)d_out;
    float* out_ids = out_f + 3;
    float* xrec    = out_f + 3 + (size_t)cB * cL;

    const int CVT_BLKS = 2048;
    // --- conversions: activations plain, weights fused convert+transpose ---
    cvt_bf16_kernel<<<CVT_BLKS, 256, 0, stream>>>(x, xbf, (size_t)cB * cIN);
    auto cvtT = [&](const float* in, unsigned short* out, int R, int Ccol) {
        dim3 grid(Ccol / 32, R / 32);
        cvt_transpose_kernel<<<grid, 256, 0, stream>>>(in, out, R, Ccol);
    };
    cvtT(enc_w1, w1t,  cIN,  cHID);
    cvtT(enc_w2, w2t,  cHID, cHID);
    cvtT(enc_w3, w3t,  cHID, cLAT);
    cvtT(dec_w1, dw1t, cLAT, cHID);
    cvtT(dec_w2, dw2t, cHID, cHID);
    cvtT(dec_w3, dw3t, cHID, cIN);
    cvt_bf16_kernel<<<CVT_BLKS, 256, 0, stream>>>(cbooks, cbbf, (size_t)cL * cK * cLAT);
    cnorm_kernel<<<(cL * cK + 255) / 256, 256, 0, stream>>>(cbooks, cnorm, cL * cK, cLAT);
    zero_kernel<<<CVT_BLKS, 256, 0, stream>>>(qsum, (size_t)cB * cLAT);
    zero_kernel<<<1, 64, 0, stream>>>(loss, 64);

    auto gemm = [&](const unsigned short* A, const unsigned short* Wt,
                    const float* bias, float* C, int M, int N, int K) {
        dim3 grid(N / GBN, M / GBM);
        gemm_bf16_kernel<<<grid, 256, 0, stream>>>(A, Wt, bias, C, M, N, K);
    };

    // --- encoder ---
    gemm(xbf, w1t, enc_b1, gbuf, cB, cHID, cIN);
    ln_gelu_kernel<<<cB, 256, 0, stream>>>(gbuf, enc_g1, enc_be1, act, cHID);
    gemm(act, w2t, enc_b2, gbuf, cB, cHID, cHID);
    ln_gelu_kernel<<<cB, 256, 0, stream>>>(gbuf, enc_g2, enc_be2, act, cHID);
    gemm(act, w3t, enc_b3, res, cB, cLAT, cHID);   // z -> residual buffer

    // --- residual VQ (4 serialized levels) ---
    for (int l = 0; l < cL; ++l) {
        vq_level_kernel<<<cB / 64, 128, 0, stream>>>(
            l, res, qsum, cbbf, cbooks, cnorm, out_ids, loss + 0);
    }

    // --- decoder ---
    cvt_bf16_kernel<<<CVT_BLKS, 256, 0, stream>>>(qsum, qsumbf, (size_t)cB * cLAT);
    gemm(qsumbf, dw1t, dec_b1, gbuf, cB, cHID, cLAT);
    ln_gelu_kernel<<<cB, 256, 0, stream>>>(gbuf, dec_g1, dec_be1, act, cHID);
    gemm(act, dw2t, dec_b2, gbuf, cB, cHID, cHID);
    ln_gelu_kernel<<<cB, 256, 0, stream>>>(gbuf, dec_g2, dec_be2, act, cHID);
    gemm(act, dw3t, dec_b3, xrec, cB, cIN, cHID);  // x_recon straight to d_out

    // --- losses ---
    sqdiff_kernel<<<2048, 256, 0, stream>>>(xrec, x, loss + 1, (size_t)cB * cIN);
    finalize_kernel<<<1, 1, 0, stream>>>(loss, out_f);
}
